// tri_gat_60971355734197
// MI455X (gfx1250) — matrix operand order repaired and checked by reading, not compiled, not run on hardware
//
#include <hip/hip_runtime.h>
#include <hip/hip_bf16.h>

// ---------------- problem constants (match reference) ----------------
#define NN   100000   // nodes
#define EC   800000   // edges
#define BB   2500     // graphs
#define HD   64       // hidden
// HEADS*H = 256, DEG = 8 (contiguous per dst), 40 nodes per graph

typedef _Float16 v16h __attribute__((ext_vector_type(16)));
typedef _Float16 v8h  __attribute__((ext_vector_type(8)));
typedef float    v8f  __attribute__((ext_vector_type(8)));

#define WMMA_F16(A, B, C) \
  __builtin_amdgcn_wmma_f32_16x16x32_f16(false, (A), false, (B), (short)0, (C), false, false)

// ---------------- WMMA fragment helpers ----------------
// B operand, pre-packed: lane = (khalf<<4)|col ; 16 f16 contiguous per lane.
static __device__ inline v16h load_b_packed(const _Float16* packed, int nt, int kt,
                                            int numKt, int lane) {
  return *(const v16h*)(packed + (size_t)((nt * numKt + kt) * 32 + lane) * 16);
}

// A operand (16 rows x 32 K) from row-major f16; ISA 16-bit A layout:
// lanes 0-15: K = 0..7 & 16..23 ; lanes 16-31: K = 8..15 & 24..31, row = lane&15.
static __device__ inline v16h load_a_f16(const _Float16* rowbase, int ld, int lane, int kbase) {
  int row = lane & 15;
  int khalf = (lane >> 4) * 8;
  const _Float16* p = rowbase + (size_t)row * ld + kbase + khalf;
  union { v16h v; v8h h[2]; } a;
  a.h[0] = *(const v8h*)(p);
  a.h[1] = *(const v8h*)(p + 16);
  return a.v;
}

// A operand from row-major f32 with K zero-padding (kmax = valid K).
static __device__ inline v16h load_a_f32(const float* rowbase, int ld, int lane,
                                         int kbase, int kmax) {
  int row = lane & 15;
  int khalf = (lane >> 4) * 8;
  const float* p = rowbase + (size_t)row * ld;
  union { v16h v; _Float16 e[16]; } a;
#pragma unroll
  for (int j = 0; j < 8; ++j) {
    int k0 = kbase + khalf + j;
    int k1 = k0 + 16;
    a.e[j]     = (k0 < kmax) ? (_Float16)p[k0] : (_Float16)0.f;
    a.e[8 + j] = (k1 < kmax) ? (_Float16)p[k1] : (_Float16)0.f;
  }
  return a.v;
}

// D fragment scatter: col = lane&15, rows = vgpr + 8*(lane>=16)
static __device__ inline void store_d_f32(float* out16rows, int ld, int colbase,
                                          v8f d, int lane) {
  int col = lane & 15;
  int rbase = (lane >> 4) * 8;
#pragma unroll
  for (int r = 0; r < 8; ++r)
    out16rows[(size_t)(rbase + r) * ld + colbase + col] = d[r];
}
static __device__ inline void store_d_f16(_Float16* out16rows, int ld, int colbase,
                                          v8f d, int lane) {
  int col = lane & 15;
  int rbase = (lane >> 4) * 8;
#pragma unroll
  for (int r = 0; r < 8; ++r)
    out16rows[(size_t)(rbase + r) * ld + colbase + col] = (_Float16)d[r];
}

// ---------------- weight pre-pack (f32 row-major [K x Ncols] -> WMMA-B f16) ----------------
__global__ void pack_weight_kernel(const float* __restrict__ W, _Float16* __restrict__ out,
                                   int K, int Kpad, int Ncols) {
  int idx = blockIdx.x * blockDim.x + threadIdx.x;
  int total = Kpad * Ncols;
  if (idx >= total) return;
  int numKt = Kpad >> 5;
  int tile = idx >> 9;          // 512 elements per (nt,kt) tile
  int within = idx & 511;
  int lane = within >> 4;
  int j = within & 15;
  int nt = tile / numKt;
  int kt = tile - nt * numKt;
  int col = lane & 15;
  int k = kt * 32 + (lane >> 4) * 8 + (j & 7) + (j >> 3) * 16;
  float v = (k < K) ? W[(size_t)k * Ncols + nt * 16 + col] : 0.f;
  out[idx] = (_Float16)v;
}

// ---------------- projections ----------------
// hh = h[N,44] @ W_proj[44,64]  (K padded to 64)
__global__ void proj_h_kernel(const float* __restrict__ h, const _Float16* __restrict__ Wp,
                              float* __restrict__ hh) {
  int lane = threadIdx.x;
  int nb = blockIdx.x * 16;
  v16h a0 = load_a_f32(h + (size_t)nb * 44, 44, lane, 0, 44);
  v16h a1 = load_a_f32(h + (size_t)nb * 44, 44, lane, 32, 44);
#pragma unroll
  for (int nt = 0; nt < 4; ++nt) {
    v8f c = {0.f,0.f,0.f,0.f,0.f,0.f,0.f,0.f};
    c = WMMA_F16(a0, load_b_packed(Wp, nt, 0, 2, lane), c);
    c = WMMA_F16(a1, load_b_packed(Wp, nt, 1, 2, lane), c);
    store_d_f32(hh + (size_t)nb * HD, HD, nt * 16, c, lane);
  }
}

// ee = e_f[E,13] @ W_proj_e[13,64]  (K padded to 32), stored f16
__global__ void proj_e_kernel(const float* __restrict__ ef, const _Float16* __restrict__ Wpe,
                              _Float16* __restrict__ ee) {
  int lane = threadIdx.x;
  int eb = blockIdx.x * 16;
  v16h a0 = load_a_f32(ef + (size_t)eb * 13, 13, lane, 0, 13);
#pragma unroll
  for (int nt = 0; nt < 4; ++nt) {
    v8f c = {0.f,0.f,0.f,0.f,0.f,0.f,0.f,0.f};
    c = WMMA_F16(a0, load_b_packed(Wpe, nt, 0, 1, lane), c);
    store_d_f16(ee + (size_t)eb * HD, HD, nt * 16, c, lane);
  }
}

// ---------------- conv kernel 1: wv = hh @ Wn  (+ per-node dst attention dot) ----------------
__global__ void node_wv_kernel(const float* __restrict__ hh, const _Float16* __restrict__ Wn_p,
                               const float* __restrict__ atrip,  // [4][192] for this conv
                               _Float16* __restrict__ wv, float* __restrict__ dotv) {
  __shared__ _Float16 lds_wv[16 * 256];   // staging for coalesced b128 stores
  int lane = threadIdx.x;
  int nb = blockIdx.x * 16;
  v16h a0 = load_a_f32(hh + (size_t)nb * HD, HD, lane, 0, HD);
  v16h a1 = load_a_f32(hh + (size_t)nb * HD, HD, lane, 32, HD);
  int col = lane & 15;
  int rbase = (lane >> 4) * 8;
#pragma unroll
  for (int hd = 0; hd < 4; ++hd) {
    float pv[8];
#pragma unroll
    for (int r = 0; r < 8; ++r) pv[r] = 0.f;
#pragma unroll
    for (int sub = 0; sub < 4; ++sub) {
      int nt = hd * 4 + sub;
      v8f c = {0.f,0.f,0.f,0.f,0.f,0.f,0.f,0.f};
      c = WMMA_F16(a0, load_b_packed(Wn_p, nt, 0, 2, lane), c);
      c = WMMA_F16(a1, load_b_packed(Wn_p, nt, 1, 2, lane), c);
      // stage wv tile in LDS (f16)
#pragma unroll
      for (int r = 0; r < 8; ++r)
        lds_wv[(rbase + r) * 256 + nt * 16 + col] = (_Float16)c[r];
      // accumulate av . wv  (av = third chunk of a_trip[hd])
      float av = atrip[hd * 192 + 128 + sub * 16 + col];
#pragma unroll
      for (int r = 0; r < 8; ++r) pv[r] += c[r] * av;
    }
    // reduce over 16 columns (stay within each half-wave: rows differ per half)
#pragma unroll
    for (int m = 1; m < 16; m <<= 1)
#pragma unroll
      for (int r = 0; r < 8; ++r) pv[r] += __shfl_xor(pv[r], m, 32);
    if (col == 0) {
#pragma unroll
      for (int r = 0; r < 8; ++r)
        dotv[(size_t)(nb + rbase + r) * 4 + hd] = pv[r];
    }
  }
  __syncthreads();
  // coalesced flush: 16x256 f16 = 512 v8h chunks, 16 per lane
  _Float16* gout = wv + (size_t)nb * 256;
#pragma unroll
  for (int i = 0; i < 16; ++i) {
    int chunk = i * 32 + lane;
    *(v8h*)(gout + chunk * 8) = *(const v8h*)(lds_wv + chunk * 8);
  }
}

// ---------------- conv kernel 2: fused edge GEMM + attention + softmax + aggregate ----------
// one wave per 2 dst nodes (16 contiguous edges)
__global__ void edge_conv_kernel(const _Float16* __restrict__ ee, const _Float16* __restrict__ We_p,
                                 const _Float16* __restrict__ wv, const float* __restrict__ dotv,
                                 const int* __restrict__ src, const float* __restrict__ atrip,
                                 _Float16* __restrict__ agg) {
  __shared__ float lds_et[16 * 256];   // et = ee_tile @ We, f32
  __shared__ float lds_sc[16 * 4];     // logits -> scores
  __shared__ float lds_md[8 * 2];      // per (node,head): max, 1/den
  int lane = threadIdx.x;
  int n0 = blockIdx.x * 2;
  long e0 = (long)n0 * 8;
  int col = lane & 15;
  int rbase = (lane >> 4) * 8;

  // prefetch the randomly-gathered u rows (512B each) so they land in-cache
  // while the WMMA phase below runs
  int el = lane & 15;
  int s = src[e0 + el];
  const _Float16* urow = wv + (size_t)s * 256;
  __builtin_prefetch(urow, 0, 1);
  __builtin_prefetch(urow + 128, 0, 1);

  // phase A: et[16x256] via WMMA, into LDS
  v16h a0 = load_a_f16(ee + (size_t)e0 * HD, HD, lane, 0);
  v16h a1 = load_a_f16(ee + (size_t)e0 * HD, HD, lane, 32);
#pragma unroll
  for (int nt = 0; nt < 16; ++nt) {
    v8f c = {0.f,0.f,0.f,0.f,0.f,0.f,0.f,0.f};
    c = WMMA_F16(a0, load_b_packed(We_p, nt, 0, 2, lane), c);
    c = WMMA_F16(a1, load_b_packed(We_p, nt, 1, 2, lane), c);
#pragma unroll
    for (int r = 0; r < 8; ++r)
      lds_et[(rbase + r) * 256 + nt * 16 + col] = c[r];
  }
  __syncthreads();

  // phase B: logits; lane handles edge el = lane&15, two heads
  int nd = el >> 3;
#pragma unroll
  for (int t = 0; t < 2; ++t) {
    int hd = (lane >> 4) * 2 + t;
    const float* au = atrip + hd * 192;       // first chunk
    const float* ae = au + 64;                // second chunk
    float acc = dotv[(size_t)(n0 + nd) * 4 + hd];
    float da = 0.f, de = 0.f;
    const v8h* up = (const v8h*)(urow + hd * 64);
#pragma unroll
    for (int v = 0; v < 8; ++v) {
      v8h uv = up[v];
#pragma unroll
      for (int j = 0; j < 8; ++j) {
        da += (float)uv[j] * au[v * 8 + j];
        de += lds_et[el * 256 + hd * 64 + v * 8 + j] * ae[v * 8 + j];
      }
    }
    acc += da + de;
    if (acc < 0.f) acc *= 0.2f;               // leaky_relu
    lds_sc[el * 4 + hd] = acc;
  }
  __syncthreads();

  // per-(node,head) softmax stats over 8 edges
  if (lane < 8) {
    int ng = lane >> 2, hg = lane & 3;
    float m = -1e30f;
#pragma unroll
    for (int j = 0; j < 8; ++j) m = fmaxf(m, lds_sc[(ng * 8 + j) * 4 + hg]);
    float den = 0.f;
#pragma unroll
    for (int j = 0; j < 8; ++j) den += __expf(lds_sc[(ng * 8 + j) * 4 + hg] - m);
    lds_md[lane * 2] = m;
    lds_md[lane * 2 + 1] = 1.f / den;
  }
  __syncthreads();
#pragma unroll
  for (int t = 0; t < 2; ++t) {
    int hd = (lane >> 4) * 2 + t;
    int g = nd * 4 + hd;
    lds_sc[el * 4 + hd] = __expf(lds_sc[el * 4 + hd] - lds_md[g * 2]) * lds_md[g * 2 + 1];
  }
  __syncthreads();

  // phase C: agg[n, feat] = sum_j score * u * et  (coalesced over lanes)
#pragma unroll
  for (int n = 0; n < 2; ++n) {
    const int* srcp = src + e0 + n * 8;
#pragma unroll
    for (int c8 = 0; c8 < 8; ++c8) {
      int feat = c8 * 32 + lane;   // 0..255
      int hd = feat >> 6;
      float accv = 0.f;
#pragma unroll
      for (int j = 0; j < 8; ++j) {
        int sj = srcp[j];
        float u = (float)wv[(size_t)sj * 256 + feat];
        float e = lds_et[(n * 8 + j) * 256 + feat];
        accv += lds_sc[(n * 8 + j) * 4 + hd] * u * e;
      }
      agg[(size_t)(n0 + n) * 256 + feat] = (_Float16)accv;
    }
  }
}

// ---------------- conv kernel 3: hh += agg @ W_scale + conv_b ----------------
__global__ void scale_res_kernel(const _Float16* __restrict__ agg, const _Float16* __restrict__ Ws_p,
                                 const float* __restrict__ convb, float* __restrict__ hh) {
  int lane = threadIdx.x;
  int nb = blockIdx.x * 16;
  v16h a[8];
#pragma unroll
  for (int kt = 0; kt < 8; ++kt)
    a[kt] = load_a_f16(agg + (size_t)nb * 256, 256, lane, kt * 32);
  int col = lane & 15;
  int rbase = (lane >> 4) * 8;
#pragma unroll
  for (int nt = 0; nt < 4; ++nt) {
    v8f c = {0.f,0.f,0.f,0.f,0.f,0.f,0.f,0.f};
#pragma unroll
    for (int kt = 0; kt < 8; ++kt)
      c = WMMA_F16(a[kt], load_b_packed(Ws_p, nt, kt, 8, lane), c);
    float b = convb[nt * 16 + col];
#pragma unroll
    for (int r = 0; r < 8; ++r)
      hh[(size_t)(nb + rbase + r) * HD + nt * 16 + col] += c[r] + b;
  }
}

// ---------------- Set2Set ----------------
__global__ void zero_kernel(float* p, int n) {
  int i = blockIdx.x * blockDim.x + threadIdx.x;
  if (i < n) p[i] = 0.f;
}

// one block (64 threads) per graph; thread t owns lane j = t of h/c
__global__ void lstm_kernel(const float* __restrict__ Wih, const float* __restrict__ Whh,
                            const float* __restrict__ bih, const float* __restrict__ bhh,
                            const float* __restrict__ qstar, float* __restrict__ hx,
                            float* __restrict__ cx) {
  int b = blockIdx.x, t = threadIdx.x;
  __shared__ float q[128];
  __shared__ float hs[64];
  q[t] = qstar[(size_t)b * 128 + t];
  q[64 + t] = qstar[(size_t)b * 128 + 64 + t];
  hs[t] = hx[(size_t)b * 64 + t];
  __syncthreads();
  float g[4];
#pragma unroll
  for (int gi = 0; gi < 4; ++gi) {
    int gr = gi * 64 + t;
    float acc = bih[gr] + bhh[gr];
    const float* wi = Wih + (size_t)gr * 128;
    const float* wh = Whh + (size_t)gr * 64;
    for (int k = 0; k < 128; ++k) acc += q[k] * wi[k];
    for (int k = 0; k < 64; ++k) acc += hs[k] * wh[k];
    g[gi] = acc;
  }
  float ig = 1.f / (1.f + __expf(-g[0]));
  float fg = 1.f / (1.f + __expf(-g[1]));
  float gg = tanhf(g[2]);
  float og = 1.f / (1.f + __expf(-g[3]));
  float c = fg * cx[(size_t)b * 64 + t] + ig * gg;
  cx[(size_t)b * 64 + t] = c;
  hx[(size_t)b * 64 + t] = og * tanhf(c);
}

// one block (64 threads) per graph: attention over its 40 contiguous nodes
__global__ void readout_kernel(const float* __restrict__ hh, const float* __restrict__ hx,
                               float* __restrict__ qstar) {
  int b = blockIdx.x, t = threadIdx.x;
  __shared__ float q[64];
  __shared__ float sc[40];
  __shared__ float md[2];
  q[t] = hx[(size_t)b * 64 + t];
  __syncthreads();
  if (t < 40) {
    const float* row = hh + (size_t)(b * 40 + t) * HD;
    float s = 0.f;
    for (int k = 0; k < 64; ++k) s += row[k] * q[k];
    sc[t] = s;
  }
  __syncthreads();
  if (t == 0) {
    float m = -1e30f;
    for (int n = 0; n < 40; ++n) m = fmaxf(m, sc[n]);
    float den = 0.f;
    for (int n = 0; n < 40; ++n) den += __expf(sc[n] - m);
    md[0] = m;
    md[1] = 1.f / den;
  }
  __syncthreads();
  if (t < 40) sc[t] = __expf(sc[t] - md[0]) * md[1];
  __syncthreads();
  float r = 0.f;
  for (int n = 0; n < 40; ++n) r += sc[n] * hh[(size_t)(b * 40 + n) * HD + t];
  qstar[(size_t)b * 128 + t] = q[t];
  qstar[(size_t)b * 128 + 64 + t] = r;
}

__global__ void out_kernel(const float* __restrict__ qstar, const float* __restrict__ Wout,
                           const float* __restrict__ bout, float* __restrict__ out) {
  int id = blockIdx.x * blockDim.x + threadIdx.x;
  if (id >= BB * 64) return;
  int b = id >> 6, j = id & 63;
  float acc = bout[j];
  const float* q = qstar + (size_t)b * 128;
  for (int k = 0; k < 128; ++k) acc += q[k] * Wout[(size_t)k * 64 + j];
  out[id] = acc;
}

// ---------------- launch ----------------
extern "C" void kernel_launch(void* const* d_in, const int* in_sizes, int n_in,
                              void* d_out, int out_size, void* d_ws, size_t ws_size,
                              hipStream_t stream) {
  const float* h      = (const float*)d_in[0];
  const float* e_f    = (const float*)d_in[1];
  const int*   src    = (const int*)d_in[2];
  // d_in[3] = dst (contiguous blocks of 8 -> implicit), d_in[4] = graph_id (implicit)
  const float* W_proj   = (const float*)d_in[5];
  const float* W_proj_e = (const float*)d_in[6];
  const float* Wn       = (const float*)d_in[7];
  const float* We       = (const float*)d_in[8];
  const float* a_trip   = (const float*)d_in[9];
  const float* W_scale  = (const float*)d_in[10];
  const float* conv_b   = (const float*)d_in[11];
  const float* W_ih     = (const float*)d_in[12];
  const float* W_hh     = (const float*)d_in[13];
  const float* b_ih     = (const float*)d_in[14];
  const float* b_hh     = (const float*)d_in[15];
  const float* W_out    = (const float*)d_in[16];
  const float* b_out    = (const float*)d_in[17];
  float* out = (float*)d_out;

  char* w = (char*)d_ws;
  auto alloc = [&](size_t bytes) -> char* {
    char* p = w;
    w += (bytes + 255) & ~(size_t)255;
    return p;
  };
  _Float16* Wproj_p  = (_Float16*)alloc((size_t)64 * 64 * 2);
  _Float16* WprojE_p = (_Float16*)alloc((size_t)32 * 64 * 2);
  _Float16* Wn_p     = (_Float16*)alloc((size_t)3 * 64 * 256 * 2);
  _Float16* We_p     = (_Float16*)alloc((size_t)3 * 64 * 256 * 2);
  _Float16* Wsc_p    = (_Float16*)alloc((size_t)3 * 256 * 64 * 2);
  float*    hh       = (float*)alloc((size_t)NN * 64 * 4);
  _Float16* ee       = (_Float16*)alloc((size_t)EC * 64 * 2);
  _Float16* wv       = (_Float16*)alloc((size_t)NN * 256 * 2);
  float*    dotv     = (float*)alloc((size_t)NN * 4 * 4);
  _Float16* agg      = (_Float16*)alloc((size_t)NN * 256 * 2);
  float*    qstar    = (float*)alloc((size_t)BB * 128 * 4);
  float*    hx       = (float*)alloc((size_t)BB * 64 * 4);
  float*    cx       = (float*)alloc((size_t)BB * 64 * 4);

  // ---- pack weights into WMMA-B f16 layout ----
  pack_weight_kernel<<<(64 * 64 + 255) / 256, 256, 0, stream>>>(W_proj, Wproj_p, 44, 64, 64);
  pack_weight_kernel<<<(32 * 64 + 255) / 256, 256, 0, stream>>>(W_proj_e, WprojE_p, 13, 32, 64);
  for (int i = 0; i < 3; ++i) {
    pack_weight_kernel<<<(64 * 256 + 255) / 256, 256, 0, stream>>>(
        Wn + (size_t)i * 64 * 256, Wn_p + (size_t)i * 64 * 256, 64, 64, 256);
    pack_weight_kernel<<<(64 * 256 + 255) / 256, 256, 0, stream>>>(
        We + (size_t)i * 64 * 256, We_p + (size_t)i * 64 * 256, 64, 64, 256);
    pack_weight_kernel<<<(256 * 64 + 255) / 256, 256, 0, stream>>>(
        W_scale + (size_t)i * 256 * 64, Wsc_p + (size_t)i * 256 * 64, 256, 256, 64);
  }

  // ---- projections ----
  proj_h_kernel<<<NN / 16, 32, 0, stream>>>(h, Wproj_p, hh);
  proj_e_kernel<<<EC / 16, 32, 0, stream>>>(e_f, WprojE_p, ee);

  // ---- 3 GAT convolutions ----
  for (int i = 0; i < 3; ++i) {
    const _Float16* wnp = Wn_p + (size_t)i * 64 * 256;
    const _Float16* wep = We_p + (size_t)i * 64 * 256;
    const _Float16* wsp = Wsc_p + (size_t)i * 256 * 64;
    const float* at = a_trip + (size_t)i * 4 * 192;
    node_wv_kernel<<<NN / 16, 32, 0, stream>>>(hh, wnp, at, wv, dotv);
    edge_conv_kernel<<<NN / 2, 32, 0, stream>>>(ee, wep, wv, dotv, src, at, agg);
    scale_res_kernel<<<NN / 16, 32, 0, stream>>>(agg, wsp, conv_b + (size_t)i * 64, hh);
  }

  // ---- Set2Set ----
  zero_kernel<<<(BB * 128 + 255) / 256, 256, 0, stream>>>(qstar, BB * 128);
  zero_kernel<<<(BB * 64 + 255) / 256, 256, 0, stream>>>(hx, BB * 64);
  zero_kernel<<<(BB * 64 + 255) / 256, 256, 0, stream>>>(cx, BB * 64);
  for (int it = 0; it < 2; ++it) {
    lstm_kernel<<<BB, 64, 0, stream>>>(W_ih, W_hh, b_ih, b_hh, qstar, hx, cx);
    readout_kernel<<<BB, 64, 0, stream>>>(hh, hx, qstar);
  }
  out_kernel<<<(BB * 64 + 255) / 256, 256, 0, stream>>>(qstar, W_out, b_out, out);
}
